// GCN_Graph_53472342835549
// MI455X (gfx1250) — compile-verified, matrix operand order checked by reading, not run on hardware
//
#include <hip/hip_runtime.h>
#include <hip/hip_bf16.h>

// ---------------------------------------------------------------------------
// GCN (2-layer) + global mean pool for MI455X (gfx1250).
// GEMMs use v_wmma_f32_16x16x32_bf16 (wave32 WMMA, f32 accumulate).
// Edge aggregation uses global_atomic_add_f32 (L2-resident: 25.6MB << 192MB L2).
// ---------------------------------------------------------------------------

typedef __attribute__((ext_vector_type(16))) __bf16 v16bf;
typedef __attribute__((ext_vector_type(8)))  float  v8f;

static __device__ __forceinline__ int gtid() {
    return blockIdx.x * blockDim.x + threadIdx.x;
}

// ---------------- small utility kernels ----------------

__global__ void k_fill(float* p, float v, int n) {
    int t = gtid();
    if (t < n) p[t] = v;
}

// deg[dst] += 1 per edge (self-loop contribution pre-filled as 1.0)
__global__ void k_deg_edges(const int* __restrict__ eidx, int E, float* __restrict__ deg) {
    int t = gtid();
    if (t < E) atomicAdd(&deg[eidx[E + t]], 1.0f);
}

__global__ void k_dinv(const float* __restrict__ deg, float* __restrict__ dinv, int n) {
    int t = gtid();
    if (t < n) {
        float d = deg[t];
        dinv[t] = (d > 0.0f) ? rsqrtf(d) : 0.0f;
    }
}

// Convert f32 weights (K x Nsrc, row-major) to bf16, zero-padded to K x Npad.
__global__ void k_convW(const float* __restrict__ W, void* __restrict__ Wbv,
                        int K, int Nsrc, int Npad) {
    __bf16* Wb = (__bf16*)Wbv;
    int t = gtid();
    if (t < K * Npad) {
        int k = t / Npad, n = t - k * Npad;
        float v = (n < Nsrc) ? W[k * Nsrc + n] : 0.0f;
        Wb[t] = (__bf16)v;
    }
}

// ---------------- WMMA GEMM: O[M x 16*NT] = (relu?)X[M x 64] @ W[64 x 16*NT] ----
// One wave computes a 16-row tile across all NT column tiles.
// A/B fragment lane layout per CDNA5 ISA 7.12.2 (16-bit, 16x32):
//   lanes 0-15 : K = {kk+0..7, kk+16..23};  lanes 16-31: K = {kk+8..15, kk+24..31}
template<int NT, bool RELU, int OSTRIDE>
__global__ void k_gemm_wmma(const float* __restrict__ X, const void* __restrict__ Wbv,
                            float* __restrict__ O, int nTiles) {
    const __bf16* Wg = (const __bf16*)Wbv;
    __shared__ __bf16 Wl[64 * 16 * NT];
    for (int i = threadIdx.x; i < 64 * 16 * NT; i += blockDim.x) Wl[i] = Wg[i];
    __syncthreads();

    const int lane = threadIdx.x & 31;
    const int wave = threadIdx.x >> 5;
    const int wpb  = blockDim.x >> 5;
    const int half = lane >> 4;     // which 16-lane half
    const int l16  = lane & 15;

    // Build B fragments once (column l16 of each N-tile, K packed per ISA layout)
    v16bf b[2][NT];
    for (int ks = 0; ks < 2; ++ks) {
        const int kk = ks * 32;
        for (int nt = 0; nt < NT; ++nt) {
            v16bf f;
#pragma unroll
            for (int j = 0; j < 16; ++j) {
                int k = kk + j + ((j >= 8) ? 8 : 0) + (half ? 8 : 0);
                f[j] = Wl[k * (16 * NT) + nt * 16 + l16];
            }
            b[ks][nt] = f;
        }
    }

    for (int t = blockIdx.x * wpb + wave; t < nTiles; t += gridDim.x * wpb) {
        const float* xr = X + (size_t)(t * 16 + l16) * 64;   // this lane's A row
        v8f c[NT] = {};
#pragma unroll
        for (int ks = 0; ks < 2; ++ks) {
            const int kk = ks * 32 + (half ? 8 : 0);
            v16bf a;
#pragma unroll
            for (int j = 0; j < 16; ++j) {
                int k = kk + j + ((j >= 8) ? 8 : 0);
                float v = xr[k];
                if (RELU) v = v > 0.0f ? v : 0.0f;
                a[j] = (__bf16)v;
            }
#pragma unroll
            for (int nt = 0; nt < NT; ++nt)
                c[nt] = __builtin_amdgcn_wmma_f32_16x16x32_bf16(
                    false, a, false, b[ks][nt], (short)0, c[nt], false, false);
        }
        // C/D layout: lanes 0-15 -> N=l16, VGPR r -> M=r; lanes 16-31 -> M=8+r.
#pragma unroll
        for (int nt = 0; nt < NT; ++nt)
#pragma unroll
            for (int r = 0; r < 8; ++r)
                O[(size_t)(t * 16 + r + 8 * half) * OSTRIDE + nt * 16 + l16] = c[nt][r];
    }
}

// ---------------- aggregation ----------------

// a[i][f] = bias[f] + h[i][f] * dinv[i]^2   (self-loop term), zero in pad cols
__global__ void k_selfbias(const float* __restrict__ h, const float* __restrict__ dinv,
                           const float* __restrict__ bias, float* __restrict__ a,
                           int N, int F, int Fvalid) {
    int t = gtid();
    if (t < N * F) {
        int i = t / F, f = t - i * F;
        float di = dinv[i];
        a[t] = (f < Fvalid) ? bias[f] + h[t] * di * di : 0.0f;
    }
}

// Layer-1 scatter: 16 threads/edge, float4 gather + 4 f32 atomics each (64 feats)
__global__ void k_edge_agg64(const int* __restrict__ eidx, int E,
                             const float* __restrict__ dinv,
                             const float* __restrict__ h, float* __restrict__ a) {
    int t = gtid();
    if (t >= E * 16) return;
    int e = t >> 4, f = (t & 15) << 2;
    int s = eidx[e], d = eidx[E + e];
    float nrm = dinv[s] * dinv[d];
    float4 v = *(const float4*)(h + (size_t)s * 64 + f);
    float* ap = a + (size_t)d * 64 + f;
    atomicAdd(ap + 0, v.x * nrm);
    atomicAdd(ap + 1, v.y * nrm);
    atomicAdd(ap + 2, v.z * nrm);
    atomicAdd(ap + 3, v.w * nrm);
}

// Layer-2 scatter: 10 threads/edge (10 classes, stride-16 storage)
__global__ void k_edge_agg10(const int* __restrict__ eidx, int E,
                             const float* __restrict__ dinv,
                             const float* __restrict__ h, float* __restrict__ a) {
    int t = gtid();
    if (t >= E * 10) return;
    int e = t / 10, c = t - e * 10;
    int s = eidx[e], d = eidx[E + e];
    float nrm = dinv[s] * dinv[d];
    atomicAdd(&a[(size_t)d * 16 + c], h[(size_t)s * 16 + c] * nrm);
}

// ---------------- pooling ----------------

__global__ void k_count(const int* __restrict__ batch, float* __restrict__ counts, int N) {
    int t = gtid();
    if (t < N) atomicAdd(&counts[batch[t]], 1.0f);
}

__global__ void k_poolsum(const int* __restrict__ batch, const float* __restrict__ a,
                          float* __restrict__ sums, int N) {
    int t = gtid();
    if (t >= N * 10) return;
    int n = t / 10, c = t - n * 10;
    atomicAdd(&sums[batch[n] * 10 + c], a[(size_t)n * 16 + c]);
}

__global__ void k_final(const float* __restrict__ sums, const float* __restrict__ counts,
                        float* __restrict__ out, int total) {
    int t = gtid();
    if (t < total) {
        int g = t / 10;
        out[t] = sums[t] / fmaxf(counts[g], 1.0f);
    }
}

// ---------------------------------------------------------------------------

extern "C" void kernel_launch(void* const* d_in, const int* in_sizes, int n_in,
                              void* d_out, int out_size, void* d_ws, size_t ws_size,
                              hipStream_t stream) {
    const float* x    = (const float*)d_in[0];
    const int*   edge = (const int*)d_in[1];   // [2,E] flat: src row then dst row
    const int*   batch= (const int*)d_in[2];
    const float* W1   = (const float*)d_in[3];
    const float* b1   = (const float*)d_in[4];
    const float* W2   = (const float*)d_in[5];
    const float* b2   = (const float*)d_in[6];
    float* out = (float*)d_out;

    const int N = in_sizes[2];        // 100000
    const int E = in_sizes[1] / 2;    // 1600000
    const int nTiles = N / 16;        // 6250 (exact)

    // workspace carve-out
    char* p = (char*)d_ws;
    auto alloc = [&](size_t bytes) -> char* {
        char* r = p;
        p += (bytes + 255) & ~(size_t)255;
        return r;
    };
    float* deg   = (float*)alloc((size_t)N * 4);
    float* dinv  = (float*)alloc((size_t)N * 4);
    void*  Wb1   = (void*)alloc(64 * 64 * 2);           // bf16 64x64
    void*  Wb2   = (void*)alloc(64 * 16 * 2);           // bf16 64x16 (10 valid cols)
    float* h1    = (float*)alloc((size_t)N * 64 * 4);   // x @ W1
    float* a1    = (float*)alloc((size_t)N * 64 * 4);   // aggregated layer 1
    float* h2    = (float*)alloc((size_t)N * 16 * 4);   // relu(a1) @ W2
    float* sums  = (float*)alloc(128 * 10 * 4);
    float* counts= (float*)alloc(128 * 4);
    float* a2    = h1;                                  // h1 dead after GEMM2; reuse

    const int T = 256;
    auto blk = [](long long n, int t) { return (unsigned)((n + t - 1) / t); };

    // 1) degree (self-loop = 1.0 baseline) and symmetric norm
    k_fill<<<blk(N, T), T, 0, stream>>>(deg, 1.0f, N);
    k_deg_edges<<<blk(E, T), T, 0, stream>>>(edge, E, deg);
    k_dinv<<<blk(N, T), T, 0, stream>>>(deg, dinv, N);

    // 2) weights -> bf16 (padded)
    k_convW<<<blk(64 * 64, T), T, 0, stream>>>(W1, Wb1, 64, 64, 64);
    k_convW<<<blk(64 * 16, T), T, 0, stream>>>(W2, Wb2, 64, 10, 16);

    // 3) layer 1: h1 = x @ W1  (WMMA)
    k_gemm_wmma<4, false, 64><<<blk(nTiles, 4), 128, 0, stream>>>(x, Wb1, h1, nTiles);

    // 4) layer-1 aggregation: a1 = b1 + selfloop + scatter-add over edges
    k_selfbias<<<blk((long long)N * 64, T), T, 0, stream>>>(h1, dinv, b1, a1, N, 64, 64);
    k_edge_agg64<<<blk((long long)E * 16, T), T, 0, stream>>>(edge, E, dinv, h1, a1);

    // 5) layer 2: h2 = relu(a1) @ W2  (WMMA, relu fused into A-frag conversion)
    k_gemm_wmma<1, true, 16><<<blk(nTiles, 4), 128, 0, stream>>>(a1, Wb2, h2, nTiles);

    // 6) layer-2 aggregation
    k_selfbias<<<blk((long long)N * 16, T), T, 0, stream>>>(h2, dinv, b2, a2, N, 16, 10);
    k_edge_agg10<<<blk((long long)E * 10, T), T, 0, stream>>>(edge, E, dinv, h2, a2);

    // 7) global mean pool
    k_fill<<<blk(128 * 10, T), T, 0, stream>>>(sums, 0.0f, 128 * 10);
    k_fill<<<blk(128, T), T, 0, stream>>>(counts, 0.0f, 128);
    k_count<<<blk(N, T), T, 0, stream>>>(batch, counts, N);
    k_poolsum<<<blk((long long)N * 10, T), T, 0, stream>>>(batch, a2, sums, N);
    k_final<<<blk(128 * 10, T), T, 0, stream>>>(sums, counts, out, 128 * 10);
}
